// ParallelBinaryBasicBlock_77025943486794
// MI455X (gfx1250) — 8-bit matrix operand order repaired and checked by reading, compiled, not run on hardware
//
#include <hip/hip_runtime.h>
#include <cstdint>

typedef __attribute__((ext_vector_type(8))) int v8i;

#define HWP    784
#define KDIM   2304

__global__ void bn_prep_kernel(const float* __restrict__ s1, const float* __restrict__ b1,
                               const float* __restrict__ m1, const float* __restrict__ v1,
                               const float* __restrict__ s2, const float* __restrict__ b2,
                               const float* __restrict__ m2, const float* __restrict__ v2,
                               float* __restrict__ inv1, float* __restrict__ beta1,
                               float* __restrict__ inv2, float* __restrict__ beta2) {
  int i = blockIdx.x * blockDim.x + threadIdx.x;
  if (i < 1024) {
    float iv = s1[i] * rsqrtf(v1[i] + 1e-5f);
    inv1[i] = iv;  beta1[i] = b1[i] - m1[i] * iv;
    float iw = s2[i] * rsqrtf(v2[i] + 1e-5f);
    inv2[i] = iw;  beta2[i] = b2[i] - m2[i] * iw;
  }
}

__global__ void wperm_kernel(const float* __restrict__ w, unsigned* __restrict__ out, int ndw) {
  int j = blockIdx.x * blockDim.x + threadIdx.x;
  if (j >= ndw) return;
  int o   = j / 576;
  int rr  = j - o * 576;
  int khw = rr >> 6;
  int c0  = (rr & 63) * 4;
  unsigned pack = 0;
#pragma unroll
  for (int k = 0; k < 4; ++k) {
    float v = w[((size_t)o * 256 + c0 + k) * 9 + khw];
    unsigned s = (unsigned)(uint8_t)(int8_t)((v > 0.f) - (v < 0.f));
    pack |= s << (8 * k);
  }
  out[j] = pack;
}

__global__ void xpad_kernel(const float* __restrict__ x, unsigned* __restrict__ out, int ndw) {
  int j = blockIdx.x * blockDim.x + threadIdx.x;
  if (j >= ndw) return;
  int c0  = (j & 63) * 4;
  int pix = j >> 6;
  int xq  = pix % 30;
  int t   = pix / 30;
  int yq  = t % 30;
  int b   = t / 30;
  unsigned pack = 0;
  if (yq >= 1 && yq <= 28 && xq >= 1 && xq <= 28) {
#pragma unroll
    for (int k = 0; k < 4; ++k) {
      float v = x[((size_t)b * 256 + c0 + k) * HWP + (yq - 1) * 28 + (xq - 1)];
      unsigned s = (unsigned)(uint8_t)(int8_t)((v > 0.f) - (v < 0.f));
      pack |= s << (8 * k);
    }
  }
  out[j] = pack;
}

__global__ void zero_kernel(int* __restrict__ p, int n) {
  int i = blockIdx.x * blockDim.x + threadIdx.x;
  if (i < n) p[i] = 0;
}

__global__ __launch_bounds__(256)
void bconv1_kernel(const int8_t* __restrict__ W1p, const int8_t* __restrict__ Xbp,
                   const float* __restrict__ inv1, const float* __restrict__ beta1,
                   int8_t* __restrict__ act2p) {
  __shared__ int Bpack[2][4][256];
  const int tid  = threadIdx.x;
  const int wave = tid >> 5;
  const int lane = tid & 31;
  const int half = (tid >> 4) & 1;
  const int nl   = tid & 15;

  const int K0 = (wave >> 1) * 16 + (wave & 1) * 4 + half * 8;

  const int8_t* Bb[4];
  int pb[4], poh[4], pow_[4];
#pragma unroll
  for (int ns = 0; ns < 4; ++ns) {
    const int nG  = blockIdx.x * 64 + ns * 16 + nl;
    const int b   = nG / HWP;
    const int rem = nG - b * HWP;
    const int oh  = rem / 28;
    const int ow  = rem - oh * 28;
    pb[ns] = b;  poh[ns] = oh;  pow_[ns] = ow;
    Bb[ns] = Xbp + ((size_t)((b * 30 + oh) * 30 + ow)) * 256 + K0;
  }

  const int mrow0 = blockIdx.y * 256 + wave * 32 + nl;
  const int8_t* Ab0 = W1p + (size_t)mrow0 * KDIM + half * 8;
  const int8_t* Ab1 = Ab0 + (size_t)16 * KDIM;

  v8i acc[2][4];
#pragma unroll
  for (int ms = 0; ms < 2; ++ms)
#pragma unroll
    for (int ns = 0; ns < 4; ++ns) acc[ms][ns] = v8i{};

  for (int khw = 0; khw < 9; ++khw) {
    const int kh   = khw / 3;
    const int kw   = khw - kh * 3;
    const int toff = (kh * 30 + kw) * 256;
    const int aoff = khw * 256;
#pragma unroll
    for (int sub = 0; sub < 4; ++sub) {
      const int buf = sub & 1;
#pragma unroll
      for (int ns = 0; ns < 4; ++ns) {
        const unsigned bw = *reinterpret_cast<const unsigned*>(Bb[ns] + toff + sub * 64);
        Bpack[buf][ns][tid] = (int)bw;
      }
      __syncthreads();
      v8i a0, a1;
#pragma unroll
      for (int p = 0; p < 4; ++p) {
        const int2 t0 = *reinterpret_cast<const int2*>(Ab0 + aoff + sub * 64 + p * 16);
        const int2 t1 = *reinterpret_cast<const int2*>(Ab1 + aoff + sub * 64 + p * 16);
        a0[2 * p] = t0.x;  a0[2 * p + 1] = t0.y;
        a1[2 * p] = t1.x;  a1[2 * p + 1] = t1.y;
      }
#pragma unroll
      for (int ns = 0; ns < 4; ++ns) {
        v8i bfr;
#pragma unroll
        for (int v = 0; v < 8; ++v) bfr[v] = Bpack[buf][ns][v * 32 + lane];
        acc[0][ns] = __builtin_amdgcn_wmma_i32_16x16x64_iu8(true, a0, true, bfr,
                                                            acc[0][ns], false, false);
        acc[1][ns] = __builtin_amdgcn_wmma_i32_16x16x64_iu8(true, a1, true, bfr,
                                                            acc[1][ns], false, false);
      }
    }
  }

#pragma unroll
  for (int ms = 0; ms < 2; ++ms) {
    const int ocs = blockIdx.y * 256 + wave * 32 + ms * 16 + half * 8;
    const float4 i0 = *reinterpret_cast<const float4*>(inv1 + ocs);
    const float4 i1 = *reinterpret_cast<const float4*>(inv1 + ocs + 4);
    const float4 e0 = *reinterpret_cast<const float4*>(beta1 + ocs);
    const float4 e1 = *reinterpret_cast<const float4*>(beta1 + ocs + 4);
    const float iv[8] = {i0.x, i0.y, i0.z, i0.w, i1.x, i1.y, i1.z, i1.w};
    const float be[8] = {e0.x, e0.y, e0.z, e0.w, e1.x, e1.y, e1.z, e1.w};
#pragma unroll
    for (int ns = 0; ns < 4; ++ns) {
      unsigned lo = 0, hi = 0;
#pragma unroll
      for (int r = 0; r < 4; ++r)
        if ((float)acc[ms][ns][r] * iv[r] + be[r] > 0.f) lo |= 1u << (8 * r);
#pragma unroll
      for (int r = 4; r < 8; ++r)
        if ((float)acc[ms][ns][r] * iv[r] + be[r] > 0.f) hi |= 1u << (8 * (r - 4));
      int2 st;  st.x = (int)lo;  st.y = (int)hi;
      *reinterpret_cast<int2*>(act2p +
          ((size_t)((pb[ns] * 30 + poh[ns] + 1) * 30 + (pow_[ns] + 1))) * 1024 + ocs) = st;
    }
  }
}

__global__ __launch_bounds__(256)
void bconv2_kernel(const int8_t* __restrict__ W2p, const int8_t* __restrict__ act2p,
                   const float* __restrict__ x,
                   const float* __restrict__ inv2, const float* __restrict__ beta2,
                   const float* __restrict__ lambdas, float* __restrict__ out) {
  __shared__ int Bpack[2][4][256];
  const int tid  = threadIdx.x;
  const int wave = tid >> 5;
  const int lane = tid & 31;
  const int half = (tid >> 4) & 1;
  const int nl   = tid & 15;

  const int K0 = (wave >> 1) * 16 + (wave & 1) * 4 + half * 8;

  const int8_t* Bb[4];
  int pb[4], poh[4], pow_[4];
#pragma unroll
  for (int ns = 0; ns < 4; ++ns) {
    const int nG  = blockIdx.x * 64 + ns * 16 + nl;
    const int b   = nG / HWP;
    const int rem = nG - b * HWP;
    const int oh  = rem / 28;
    const int ow  = rem - oh * 28;
    pb[ns] = b;  poh[ns] = oh;  pow_[ns] = ow;
    Bb[ns] = act2p + ((size_t)((b * 30 + oh) * 30 + ow)) * 1024 + K0;
  }

  const int mrow0 = wave * 32 + nl;
  const int ocs0  = wave * 32 + half * 8;

  float facc[2][4][8];
#pragma unroll
  for (int ms = 0; ms < 2; ++ms)
#pragma unroll
    for (int ns = 0; ns < 4; ++ns)
#pragma unroll
      for (int r = 0; r < 8; ++r) facc[ms][ns][r] = 0.f;

  for (int g = 0; g < 4; ++g) {
    const int8_t* Ab0 = W2p + (size_t)(g * 256 + mrow0) * KDIM + half * 8;
    const int8_t* Ab1 = Ab0 + (size_t)16 * KDIM;
    v8i acc[2][4];
#pragma unroll
    for (int ms = 0; ms < 2; ++ms)
#pragma unroll
      for (int ns = 0; ns < 4; ++ns) acc[ms][ns] = v8i{};

    for (int khw = 0; khw < 9; ++khw) {
      const int kh   = khw / 3;
      const int kw   = khw - kh * 3;
      const int toff = (kh * 30 + kw) * 1024 + g * 256;
      const int aoff = khw * 256;
#pragma unroll
      for (int sub = 0; sub < 4; ++sub) {
        const int buf = sub & 1;
#pragma unroll
        for (int ns = 0; ns < 4; ++ns) {
          const unsigned bw = *reinterpret_cast<const unsigned*>(Bb[ns] + toff + sub * 64);
          Bpack[buf][ns][tid] = (int)bw;
        }
        __syncthreads();
        v8i a0, a1;
#pragma unroll
        for (int p = 0; p < 4; ++p) {
          const int2 t0 = *reinterpret_cast<const int2*>(Ab0 + aoff + sub * 64 + p * 16);
          const int2 t1 = *reinterpret_cast<const int2*>(Ab1 + aoff + sub * 64 + p * 16);
          a0[2 * p] = t0.x;  a0[2 * p + 1] = t0.y;
          a1[2 * p] = t1.x;  a1[2 * p + 1] = t1.y;
        }
#pragma unroll
        for (int ns = 0; ns < 4; ++ns) {
          v8i bfr;
#pragma unroll
          for (int v = 0; v < 8; ++v) bfr[v] = Bpack[buf][ns][v * 32 + lane];
          acc[0][ns] = __builtin_amdgcn_wmma_i32_16x16x64_iu8(true, a0, true, bfr,
                                                              acc[0][ns], false, false);
          acc[1][ns] = __builtin_amdgcn_wmma_i32_16x16x64_iu8(true, a1, true, bfr,
                                                              acc[1][ns], false, false);
        }
      }
    }
    const float lam = lambdas[g];
#pragma unroll
    for (int ms = 0; ms < 2; ++ms) {
      const int oc8 = g * 256 + ocs0 + ms * 16;
      const float4 i0 = *reinterpret_cast<const float4*>(inv2 + oc8);
      const float4 i1 = *reinterpret_cast<const float4*>(inv2 + oc8 + 4);
      const float4 e0 = *reinterpret_cast<const float4*>(beta2 + oc8);
      const float4 e1 = *reinterpret_cast<const float4*>(beta2 + oc8 + 4);
      const float iv[8] = {i0.x, i0.y, i0.z, i0.w, i1.x, i1.y, i1.z, i1.w};
      const float be[8] = {e0.x, e0.y, e0.z, e0.w, e1.x, e1.y, e1.z, e1.w};
#pragma unroll
      for (int ns = 0; ns < 4; ++ns)
#pragma unroll
        for (int r = 0; r < 8; ++r)
          facc[ms][ns][r] += lam * ((float)acc[ms][ns][r] * iv[r] + be[r]);
    }
  }

#pragma unroll
  for (int ms = 0; ms < 2; ++ms) {
    const int ocs = ocs0 + ms * 16;
#pragma unroll
    for (int ns = 0; ns < 4; ++ns) {
#pragma unroll
      for (int r = 0; r < 8; ++r) {
        const size_t idx = (((size_t)pb[ns] * 256 + ocs + r) * 28 + poh[ns]) * 28 + pow_[ns];
        const float v = facc[ms][ns][r] + x[idx];
        out[idx] = v > 0.f ? v : 0.f;
      }
    }
  }
}

extern "C" void kernel_launch(void* const* d_in, const int* in_sizes, int n_in,
                              void* d_out, int out_size, void* d_ws, size_t ws_size,
                              hipStream_t stream) {
  const float* x   = (const float*)d_in[0];
  const float* w1  = (const float*)d_in[1];
  const float* s1  = (const float*)d_in[2];
  const float* b1  = (const float*)d_in[3];
  const float* m1  = (const float*)d_in[4];
  const float* v1  = (const float*)d_in[5];
  const float* w2  = (const float*)d_in[6];
  const float* s2  = (const float*)d_in[7];
  const float* b2  = (const float*)d_in[8];
  const float* m2  = (const float*)d_in[9];
  const float* v2  = (const float*)d_in[10];
  const float* lam = (const float*)d_in[11];

  int8_t* ws    = (int8_t*)d_ws;
  float*  inv1  = (float*)(ws);
  float*  beta1 = (float*)(ws + 4096);
  float*  inv2  = (float*)(ws + 8192);
  float*  beta2 = (float*)(ws + 12288);
  int8_t* W1p   = ws + 16384;
  int8_t* W2p   = ws + 2375680;
  int8_t* Xbp   = ws + 4734976;
  int8_t* act2p = ws + 12107776;

  bn_prep_kernel<<<4, 256, 0, stream>>>(s1, b1, m1, v1, s2, b2, m2, v2,
                                        inv1, beta1, inv2, beta2);

  const int wdw = 2359296 / 4;
  wperm_kernel<<<(wdw + 255) / 256, 256, 0, stream>>>(w1, (unsigned*)W1p, wdw);
  wperm_kernel<<<(wdw + 255) / 256, 256, 0, stream>>>(w2, (unsigned*)W2p, wdw);

  const int xdw = 7372800 / 4;
  xpad_kernel<<<(xdw + 255) / 256, 256, 0, stream>>>(x, (unsigned*)Xbp, xdw);

  const int adw = 29491200 / 4;
  zero_kernel<<<(adw + 255) / 256, 256, 0, stream>>>((int*)act2p, adw);

  dim3 blk(256);
  dim3 g1(392, 4);
  bconv1_kernel<<<g1, blk, 0, stream>>>(W1p, Xbp, inv1, beta1, act2p);

  dim3 g2(392, 1);
  bconv2_kernel<<<g2, blk, 0, stream>>>(W2p, act2p, x, inv2, beta2, lam,
                                        (float*)d_out);
}
